// PhasorBlock_21311627723291
// MI455X (gfx1250) — matrix operand order repaired and checked by reading, not compiled, not run on hardware
//
#include <hip/hip_runtime.h>
#include <hip/hip_bf16.h>
#include <math.h>

typedef _Float16 half_t;
typedef __attribute__((ext_vector_type(16))) _Float16 v16h;
typedef __attribute__((ext_vector_type(8)))  _Float16 v8h;
typedef __attribute__((ext_vector_type(8)))  float    v8f;

#define PI_F 3.14159265358979323846f

// ---------------------------------------------------------------------------
// Generic WMMA GEMM:  C += A(f16, MxK, lda) * B(packed f16)   (C f32, ldc)
//   B pre-packed in WMMA fragment order: [kblk][ntile][lane][e] (512/blk).
//   MT m-tiles per wave share one B fragment (M % (16*MT) == 0 required).
//   Ping-pong double buffering; acc starts at 0 and C is folded in after the
//   K-loop so no C addresses stay live across it (avoids VGPR spills).
// ---------------------------------------------------------------------------
template <int MT>
__global__ void k_gemm(const half_t* __restrict__ A, const half_t* __restrict__ Bp,
                       float* __restrict__ C,
                       int M, int N, int K, int lda, int NT, int ldc,
                       int aOS, int aIS, int bOS, int bIS, int cOS, int cIS,
                       int innerCount)
{
    int lane  = threadIdx.x;
    int z     = blockIdx.z;
    int outer = z / innerCount, inner = z % innerCount;
    const half_t* Ab = A  + (size_t)outer * aOS + (size_t)inner * aIS;
    const half_t* Bb = Bp + (size_t)outer * bOS + (size_t)inner * bIS;
    float*        Cb = C  + (size_t)outer * cOS + (size_t)inner * cIS;

    int mt0 = (blockIdx.y * 4 + threadIdx.y) * MT;
    if (mt0 * 16 >= M) return;                      // wave-uniform, outside loop

    int nt = blockIdx.x;
    int hi = (lane >= 16) ? 1 : 0;
    int n  = nt * 16 + (lane & 15);

    v8f acc[MT];
#pragma unroll
    for (int t = 0; t < MT; t++)
#pragma unroll
        for (int r = 0; r < 8; r++) acc[t][r] = 0.f;

    const half_t* ap[MT];
#pragma unroll
    for (int t = 0; t < MT; t++)
        ap[t] = Ab + (size_t)(mt0 * 16 + t * 16 + (lane & 15)) * lda + (hi ? 16 : 0);
    const half_t* bptr = Bb + (size_t)nt * 512 + (size_t)lane * 16;
    size_t bstep = (size_t)NT * 512;

    v16h a0[MT], a1[MT], b0, b1;

    auto LA = [&](v16h (&dst)[MT], int off) {
#pragma unroll
        for (int t = 0; t < MT; t++) {
            v8h lo = *(const v8h*)(ap[t] + off);
            v8h hh = *(const v8h*)(ap[t] + off + 8);
            dst[t] = __builtin_shufflevector(lo, hh, 0,1,2,3,4,5,6,7,
                                             8,9,10,11,12,13,14,15);
        }
    };
    auto WM = [&](v16h (&af)[MT], v16h bf) {
#pragma unroll
        for (int t = 0; t < MT; t++)
            acc[t] = __builtin_amdgcn_wmma_f32_16x16x32_f16(false, af[t], false, bf,
                                                            (short)0, acc[t], false, false);
    };

    int rem = K >> 5;                               // number of 32-wide k-steps
    LA(a0, 0);
    b0 = *(const v16h*)bptr;
    while (rem >= 2) {
        b1 = *(const v16h*)(bptr + bstep);
        LA(a1, 32);
        __builtin_prefetch(bptr + 2 * bstep, 0, 1);
        WM(a0, b0);
        if (rem > 2) {                              // preload set 0 for next pair
            b0 = *(const v16h*)(bptr + 2 * bstep);
            LA(a0, 64);
        }
        WM(a1, b1);
        bptr += 2 * bstep;
#pragma unroll
        for (int t = 0; t < MT; t++) ap[t] += 64;
        rem -= 2;
    }
    if (rem & 1) WM(a0, b0);                        // odd tail (K == 32 case)

    // epilogue: fold existing C (bias / prior GEMM partials) and store
    int mrow0 = mt0 * 16 + (hi ? 8 : 0);
#pragma unroll
    for (int t = 0; t < MT; t++)
#pragma unroll
        for (int r = 0; r < 8; r++) {
            float* cp = &Cb[(size_t)(mrow0 + t * 16 + r) * ldc + n];
            *cp = *cp + acc[t][r];
        }
}

// ---------------------------------------------------------------------------
// Pack weights (f32 [N][K] row-major source) into WMMA-B fragment order f16.
//   dst[i], i = ((kblk*NT + ntile)*32 + lane)*16 + e
// ---------------------------------------------------------------------------
__global__ void k_pack_w(const float* __restrict__ src, half_t* __restrict__ dst,
                         int NT, int Nsrc, int sstride, int soff, long n)
{
    for (long i = blockIdx.x * (long)blockDim.x + threadIdx.x; i < n;
         i += (long)gridDim.x * blockDim.x) {
        int  e    = (int)(i & 15);
        int  lane = (int)((i >> 4) & 31);
        long t    = i >> 9;
        int  nt   = (int)(t % NT);
        long kblk = t / NT;
        int  nn   = nt * 16 + (lane & 15);
        long k    = kblk * 32 + ((lane >= 16) ? 16 : 0) + e;
        float v = (nn < Nsrc) ? src[(size_t)nn * sstride + soff + k] : 0.f;
        dst[i] = (half_t)v;
    }
}

// Pack f32 activation [b][R][Nfull] into per-b WMMA-B fragment order f16.
__global__ void k_pack_act(const float* __restrict__ src, half_t* __restrict__ dst,
                           int NT, long perB, long n)
{
    for (long i = blockIdx.x * (long)blockDim.x + threadIdx.x; i < n;
         i += (long)gridDim.x * blockDim.x) {
        long b = i / perB, r = i % perB;
        int  e    = (int)(r & 15);
        int  lane = (int)((r >> 4) & 31);
        long t    = r >> 9;
        int  nt   = (int)(t % NT);
        long kblk = t / NT;
        int  nn   = nt * 16 + (lane & 15);
        long k    = kblk * 32 + ((lane >= 16) ? 16 : 0) + e;
        dst[i] = (half_t)src[b * perB + k * ((long)NT * 16) + nn];
    }
}

// ---------------------------------------------------------------------------
// Elementwise / prep kernels (grid-stride)
// ---------------------------------------------------------------------------
__global__ void k_convert_f16(const float* __restrict__ s, half_t* __restrict__ d, long n)
{
    for (long i = blockIdx.x * (long)blockDim.x + threadIdx.x; i < n;
         i += (long)gridDim.x * blockDim.x)
        d[i] = (half_t)s[i];
}

__global__ void k_biasinit(float* __restrict__ C, const float* __restrict__ bias,
                           int Ndst, int Nsrc, int ldc, long n)
{
    for (long i = blockIdx.x * (long)blockDim.x + threadIdx.x; i < n;
         i += (long)gridDim.x * blockDim.x) {
        long m = i / Ndst; int nn = (int)(i % Ndst);
        C[(size_t)m * ldc + nn] = (nn < Nsrc) ? bias[nn] : 0.f;
    }
}

__global__ void k_gelu(float* __restrict__ buf, half_t* __restrict__ h, long n)
{
    for (long i = blockIdx.x * (long)blockDim.x + threadIdx.x; i < n;
         i += (long)gridDim.x * blockDim.x) {
        float v = buf[i];
        float g = 0.5f * v * (1.f + erff(v * 0.70710678118654752f));
        buf[i] = g;
        if (h) h[i] = (half_t)g;
    }
}

__global__ void k_tanhpi(float* __restrict__ buf, long n)
{
    for (long i = blockIdx.x * (long)blockDim.x + threadIdx.x; i < n;
         i += (long)gridDim.x * blockDim.x)
        buf[i] = tanhf(buf[i]) * PI_F;
}

// pos trig tables: posC/posS f32 (L,P); posCt/posSt f16 transposed (P,L) [A-matrix]
__global__ void k_prep_pos(const float* __restrict__ pp, float* __restrict__ posC,
                           float* __restrict__ posS, half_t* __restrict__ posCt,
                           half_t* __restrict__ posSt, long n)
{
    for (long i = blockIdx.x * (long)blockDim.x + threadIdx.x; i < n;
         i += (long)gridDim.x * blockDim.x) {
        int l = (int)(i / 32), p = (int)(i % 32);
        float a = pp[i];
        float s, c; sincosf(a, &s, &c);
        posC[i] = c; posS[i] = s;
        posCt[(size_t)p * 2048 + l] = (half_t)c;
        posSt[(size_t)p * 2048 + l] = (half_t)s;
    }
}

__global__ void k_prep_pph(const float* __restrict__ pp, half_t* __restrict__ pph, long n)
{
    for (long i = blockIdx.x * (long)blockDim.x + threadIdx.x; i < n;
         i += (long)gridDim.x * blockDim.x) {
        int p = (int)(i % 32), l = (int)((i / 32) % 2048);
        pph[i] = (half_t)pp[(size_t)l * 32 + p];
    }
}

__global__ void k_storegate(const float* __restrict__ sgh, const float* __restrict__ w2,
                            const float* __restrict__ b2, float* __restrict__ sg, long n)
{
    for (long i = blockIdx.x * (long)blockDim.x + threadIdx.x; i < n;
         i += (long)gridDim.x * blockDim.x) {
        float acc = b2[0];
        const float* row = sgh + (size_t)i * 128;
#pragma unroll 8
        for (int j = 0; j < 128; j++) acc = fmaf(row[j], w2[j], acc);
        sg[i] = 1.f / (1.f + expf(-acc));
    }
}

__global__ void k_gatecum(const float* __restrict__ sg, float* __restrict__ gc)
{
    if (threadIdx.x != 0) return;
    int b = blockIdx.x;
    float run = 0.f;
    for (int l = 0; l < 2048; l++) {
        run += sg[(size_t)b * 2048 + l];
        gc[(size_t)b * 2048 + l] = fmaxf(run, 1.0f);
    }
}

__global__ void k_gatedv(const float* __restrict__ values, const float* __restrict__ sg,
                         float* __restrict__ gv, long n)
{
    for (long i = blockIdx.x * (long)blockDim.x + threadIdx.x; i < n;
         i += (long)gridDim.x * blockDim.x)
        gv[i] = values[i] * sg[i >> 8];
}

// ctx running mean -> f16 (A-matrix use, row-major kept)
__global__ void k_ctxavg(const float* __restrict__ ctx, half_t* __restrict__ dst, long n)
{
    for (long i = blockIdx.x * (long)blockDim.x + threadIdx.x; i < n;
         i += (long)gridDim.x * blockDim.x) {
        int b = (int)(i / 256), d = (int)(i % 256);
        float run = 0.f;
        for (int l = 0; l < 2048; l++) {
            size_t o = ((size_t)b * 2048 + l) * 256 + d;
            run += ctx[o];
            dst[o] = (half_t)(run / (float)(l + 1));
        }
    }
}

// kv A matrices: kvCt/kvSt[(b*32+p)*2048 + l] = f16(cos/sin(sp))
__global__ void k_kva(const float* __restrict__ sp, half_t* __restrict__ kvCt,
                      half_t* __restrict__ kvSt, long n)
{
    for (long i = blockIdx.x * (long)blockDim.x + threadIdx.x; i < n;
         i += (long)gridDim.x * blockDim.x) {
        int p = (int)(i % 32), l = (int)((i / 32) % 2048), b = (int)(i / (2048 * 32));
        float s, c; sincosf(sp[i], &s, &c);
        kvCt[((size_t)b * 32 + p) * 2048 + l] = (half_t)c;
        kvSt[((size_t)b * 32 + p) * 2048 + l] = (half_t)s;
    }
}

// Fold 5-delta window + 4 heads into one query pair per (b,l,p)
__global__ void k_query(const float* __restrict__ off, const float* __restrict__ posC,
                        const float* __restrict__ posS, float* __restrict__ Qc,
                        float* __restrict__ Qs, long n)
{
    for (long i = blockIdx.x * (long)blockDim.x + threadIdx.x; i < n;
         i += (long)gridDim.x * blockDim.x) {
        int p = (int)(i % 32), l = (int)((i / 32) % 2048), b = (int)(i / (2048 * 32));
        float pc = posC[(size_t)l * 32 + p], ps = posS[(size_t)l * 32 + p];
        const float* base = off + ((size_t)b * 2048 + l) * 192;
        float qcs = 0.f, qss = 0.f;
#pragma unroll
        for (int h = 0; h < 4; h++) {
            const float* orow = base + h * 48;
            float ce  = tanhf(orow[p]) * PI_F;
            float wl  = orow[32];
            float wwv = 2.f / (1.f + expf(-wl)) + 0.1f;
            float den = wwv * wwv + 1e-6f;
            float qch = 0.f, qsh = 0.f, tot = 0.f;
#pragma unroll
            for (int dd = -2; dd <= 2; dd++) {
                float wgt = expf(-0.5f * (float)(dd * dd) / den);
                float op = ce + (float)dd * 0.1f;
                float os, oc; sincosf(op, &os, &oc);
                qch += wgt * (pc * oc - ps * os);
                qsh += wgt * (ps * oc + pc * os);
                tot += wgt;
            }
            float s = 1.f / ((tot + 1e-6f) * 4.f * 5.65685424949238f); // H*sqrt(P)
            qcs += qch * s; qss += qsh * s;
        }
        Qc[i] = qcs; Qs[i] = qss;
    }
}

// exclusive scan over chunk sums: buf[B][Cn][P][D]
__global__ void k_exclscan(float* __restrict__ buf, int Cn, long n)
{
    for (long i = blockIdx.x * (long)blockDim.x + threadIdx.x; i < n;
         i += (long)gridDim.x * blockDim.x) {
        int b = (int)(i / 8192); int r = (int)(i % 8192);
        float run = 0.f;
        for (int c = 0; c < Cn; c++) {
            size_t o = ((size_t)b * Cn + c) * 8192 + r;
            float t = buf[o]; buf[o] = run; run += t;
        }
    }
}

// Fused mem1 scan+retrieve: state (32 cos + 32 sin) per lane, queries via LDS
__global__ void k_mem1scan(const float* __restrict__ content, const float* __restrict__ posC,
                           const float* __restrict__ posS, const float* __restrict__ Qc,
                           const float* __restrict__ Qs, const float* __restrict__ Scex,
                           const float* __restrict__ Ssex, float* __restrict__ outb,
                           int L_, int Lc, int Cn)
{
    int c = blockIdx.x, b = blockIdx.y, d = threadIdx.x;
    float sc[32], ss[32];
    size_t sb = (((size_t)b * Cn + c) * 32) * 256 + d;
#pragma unroll
    for (int p = 0; p < 32; p++) { sc[p] = Scex[sb + (size_t)p * 256];
                                   ss[p] = Ssex[sb + (size_t)p * 256]; }
    __shared__ float qc[32], qs[32], pc[32], ps[32];
    int l0 = c * Lc;
    for (int li = 0; li < Lc; li++) {
        int l = l0 + li;
        __syncthreads();
        if (d < 32) {
            size_t o = (size_t)l * 32 + d;
            qc[d] = Qc[(size_t)b * L_ * 32 + o];
            qs[d] = Qs[(size_t)b * L_ * 32 + o];
            pc[d] = posC[o]; ps[d] = posS[o];
        }
        __syncthreads();
        size_t ro = ((size_t)b * L_ + l) * 256 + d;
        float cv = content[ro];
        if (li + 1 < Lc) __builtin_prefetch(&content[ro + 256], 0, 1);
        float acc = 0.f;
#pragma unroll
        for (int p = 0; p < 32; p++) {
            sc[p] = fmaf(pc[p], cv, sc[p]);
            ss[p] = fmaf(ps[p], cv, ss[p]);
            acc = fmaf(qc[p], sc[p], acc);
            acc = fmaf(qs[p], ss[p], acc);
        }
        outb[ro] = acc;
    }
}

// Fused kv scan+retrieve; accumulates into combined
__global__ void k_kvscan(const float* __restrict__ gv, const float* __restrict__ sp,
                         const float* __restrict__ keyph, const float* __restrict__ gcum,
                         const float* __restrict__ Cex, const float* __restrict__ Sex,
                         float* __restrict__ outb, int L_, int Lc, int Cn)
{
    int c = blockIdx.x, b = blockIdx.y, d = threadIdx.x;
    float skc[32], sks[32];
    size_t sb = (((size_t)b * Cn + c) * 32) * 256 + d;
#pragma unroll
    for (int p = 0; p < 32; p++) { skc[p] = Cex[sb + (size_t)p * 256];
                                   sks[p] = Sex[sb + (size_t)p * 256]; }
    __shared__ float spc[32], sps[32], kc[32], ks[32];
    int l0 = c * Lc;
    for (int li = 0; li < Lc; li++) {
        int l = l0 + li;
        __syncthreads();
        if (d < 32) {
            size_t o = ((size_t)b * L_ + l) * 32 + d;
            sincosf(sp[o], &sps[d], &spc[d]);
            sincosf(keyph[o], &ks[d], &kc[d]);
        }
        __syncthreads();
        size_t ro = ((size_t)b * L_ + l) * 256 + d;
        float gvv = gv[ro];
        if (li + 1 < Lc) __builtin_prefetch(&gv[ro + 256], 0, 1);
        float acc = 0.f;
#pragma unroll
        for (int p = 0; p < 32; p++) {
            skc[p] = fmaf(spc[p], gvv, skc[p]);
            sks[p] = fmaf(sps[p], gvv, sks[p]);
            acc = fmaf(kc[p], skc[p], acc);
            acc = fmaf(ks[p], sks[p], acc);
        }
        float sc2 = rsqrtf(gcum[(size_t)b * L_ + l]) * 0.17677669529663687f; // 1/sqrt(P)
        outb[ro] += acc * sc2;
    }
}

// LayerNorm over D=256 (block per row) -> f16 for final GEMM
__global__ void k_ln(const float* __restrict__ comb, const float* __restrict__ g,
                     const float* __restrict__ bt, half_t* __restrict__ lnh)
{
    int row = blockIdx.x, t = threadIdx.x;
    __shared__ float red[256];
    __shared__ float smu, svar;
    float v = comb[(size_t)row * 256 + t];
    red[t] = v; __syncthreads();
    for (int s = 128; s > 0; s >>= 1) { if (t < s) red[t] += red[t + s]; __syncthreads(); }
    if (t == 0) smu = red[0] * (1.f / 256.f);
    __syncthreads();
    float dv = v - smu;
    red[t] = dv * dv; __syncthreads();
    for (int s = 128; s > 0; s >>= 1) { if (t < s) red[t] += red[t + s]; __syncthreads(); }
    if (t == 0) svar = red[0] * (1.f / 256.f);
    __syncthreads();
    float ln = dv * rsqrtf(svar + 1e-5f) * g[t] + bt[t];
    lnh[(size_t)row * 256 + t] = (half_t)ln;
}

__global__ void k_final(const float* __restrict__ x, const float* __restrict__ pre,
                        float* __restrict__ out, long n)
{
    for (long i = blockIdx.x * (long)blockDim.x + threadIdx.x; i < n;
         i += (long)gridDim.x * blockDim.x)
        out[i] = x[i] + pre[i];
}

// ---------------------------------------------------------------------------
extern "C" void kernel_launch(void* const* d_in, const int* in_sizes, int n_in,
                              void* d_out, int out_size, void* d_ws, size_t ws_size,
                              hipStream_t stream)
{
    (void)in_sizes; (void)n_in; (void)out_size; (void)ws_size;
    const int Bn = 2, Ln = 2048, Hn = 4;
    const int CH = 16, CLEN = 128;          // 16 chunks of length 128
    const long BL = (long)Bn * Ln;          // 4096

    const float* x         = (const float*)d_in[0];
    const float* pos_ph    = (const float*)d_in[1];
    const float* w_content = (const float*)d_in[2];
    const float* b_content = (const float*)d_in[3];
    const float* hop_w1    = (const float*)d_in[4];
    const float* hop_b1    = (const float*)d_in[5];
    const float* hop_w2    = (const float*)d_in[6];
    const float* hop_b2    = (const float*)d_in[7];
    const float* w_cpe     = (const float*)d_in[8];
    const float* b_cpe     = (const float*)d_in[9];
    const float* w_val     = (const float*)d_in[10];
    const float* b_val     = (const float*)d_in[11];
    const float* w_ctx     = (const float*)d_in[12];
    const float* b_ctx     = (const float*)d_in[13];
    const float* kv_w1     = (const float*)d_in[14];
    const float* kv_b1     = (const float*)d_in[15];
    const float* kv_w2     = (const float*)d_in[16];
    const float* kv_b2     = (const float*)d_in[17];
    const float* sg_w1     = (const float*)d_in[18];
    const float* sg_b1     = (const float*)d_in[19];
    const float* sg_w2     = (const float*)d_in[20];
    const float* sg_b2     = (const float*)d_in[21];
    const float* ln_g      = (const float*)d_in[22];
    const float* ln_b      = (const float*)d_in[23];
    const float* w_out     = (const float*)d_in[24];
    const float* b_out     = (const float*)d_in[25];
    float* out = (float*)d_out;

    // ---- workspace arena ----
    char* ws = (char*)d_ws; size_t cur = 0;
    auto alloc = [&](size_t bytes) -> void* {
        void* p = ws + cur; cur = (cur + bytes + 255) & ~(size_t)255; return p;
    };
    half_t* xh      = (half_t*)alloc(BL * 256 * 2);
    half_t* pph     = (half_t*)alloc(BL * 32 * 2);
    float*  posC    = (float*)alloc((size_t)Ln * 32 * 4);
    float*  posS    = (float*)alloc((size_t)Ln * 32 * 4);
    half_t* posCt   = (half_t*)alloc((size_t)32 * Ln * 2);
    half_t* posSt   = (half_t*)alloc((size_t)32 * Ln * 2);
    half_t* WcT     = (half_t*)alloc(256 * 256 * 2);
    half_t* WvT     = (half_t*)alloc(256 * 256 * 2);
    half_t* WxT     = (half_t*)alloc(256 * 256 * 2);
    half_t* WoutT   = (half_t*)alloc(256 * 256 * 2);
    half_t* WcpeT   = (half_t*)alloc(256 * 32 * 2);
    half_t* Sg1T    = (half_t*)alloc(256 * 128 * 2);
    half_t* KvW1aT  = (half_t*)alloc(256 * 256 * 2);
    half_t* KvW1bT  = (half_t*)alloc(256 * 256 * 2);
    half_t* KvW2T   = (half_t*)alloc(256 * 32 * 2);
    half_t* Hop1aT  = (half_t*)alloc((size_t)Hn * 256 * 128 * 2);
    half_t* Hop1bT  = (half_t*)alloc((size_t)Hn * 32 * 128 * 2);
    half_t* Hop2T   = (half_t*)alloc((size_t)Hn * 128 * 48 * 2);
    float*  content = (float*)alloc(BL * 256 * 4);
    half_t* contenth= (half_t*)alloc(BL * 256 * 2);   // packed B
    float*  values  = (float*)alloc(BL * 256 * 4);
    float*  ctx     = (float*)alloc(BL * 256 * 4);
    half_t* ctxavgh = (half_t*)alloc(BL * 256 * 2);
    float*  sg_h    = (float*)alloc(BL * 128 * 4);
    float*  sgate   = (float*)alloc(BL * 4);
    float*  gcum    = (float*)alloc(BL * 4);
    float*  gvf     = (float*)alloc(BL * 256 * 4);
    half_t* gvh     = (half_t*)alloc(BL * 256 * 2);   // packed B
    float*  kvh     = (float*)alloc(BL * 256 * 4);
    half_t* kvhh    = (half_t*)alloc(BL * 256 * 2);
    float*  keyph   = (float*)alloc(BL * 32 * 4);
    float*  sp      = (float*)alloc(BL * 32 * 4);
    half_t* kvCt    = (half_t*)alloc((size_t)Bn * 32 * Ln * 2);
    half_t* kvSt    = (half_t*)alloc((size_t)Bn * 32 * Ln * 2);
    float*  h1      = (float*)alloc(BL * 512 * 4);
    half_t* h1h     = (half_t*)alloc(BL * 512 * 2);
    float*  offb    = (float*)alloc(BL * 192 * 4);
    float*  QcS     = (float*)alloc(BL * 32 * 4);
    float*  QsS     = (float*)alloc(BL * 32 * 4);
    float*  Sc_ch   = (float*)alloc((size_t)Bn * CH * 32 * 256 * 4);
    float*  Ss_ch   = (float*)alloc((size_t)Bn * CH * 32 * 256 * 4);
    float*  KvC_ch  = (float*)alloc((size_t)Bn * CH * 32 * 256 * 4);
    float*  KvS_ch  = (float*)alloc((size_t)Bn * CH * 32 * 256 * 4);
    float*  combined= (float*)alloc(BL * 256 * 4);
    half_t* lnh     = (half_t*)alloc(BL * 256 * 2);
    float*  outpre  = (float*)alloc(BL * 256 * 4);

    auto NB = [](long n) -> unsigned { long b = (n + 255) / 256; return (unsigned)(b < 1 ? 1 : b); };
    auto PW = [&](const float* src, half_t* dst, int Ndst, int Nsrc, int K, int ss, int so) {
        long n = (long)Ndst * K;
        k_pack_w<<<NB(n), 256, 0, stream>>>(src, dst, Ndst / 16, Nsrc, ss, so, n);
    };
    auto BI = [&](float* C, const float* bias, long M, int Ndst, int Nsrc, int ldc) {
        long n = M * Ndst;
        k_biasinit<<<NB(n), 256, 0, stream>>>(C, bias, Ndst, Nsrc, ldc, n);
    };
    // MT=4 path for the big GEMMs (M = 4096)
    auto G4 = [&](const half_t* A, const half_t* Bp, float* C, int M, int N, int K,
                  int lda, int ldc) {
        dim3 g(N / 16, (M + 255) / 256, 1), blk(32, 4);
        k_gemm<4><<<g, blk, 0, stream>>>(A, Bp, C, M, N, K, lda, N / 16, ldc,
                                         0, 0, 0, 0, 0, 0, 1);
    };
    // MT=2 batched path for the M=32 chunk GEMMs
    auto G2 = [&](const half_t* A, const half_t* Bp, float* C, int M, int N, int K,
                  int lda, int ldc, int batches, int inner,
                  int aOS, int aIS, int bOS, int bIS, int cOS, int cIS) {
        dim3 g(N / 16, (M + 127) / 128, batches), blk(32, 4);
        k_gemm<2><<<g, blk, 0, stream>>>(A, Bp, C, M, N, K, lda, N / 16, ldc,
                                         aOS, aIS, bOS, bIS, cOS, cIS, inner);
    };

    // ---- prep: f16 inputs, trig tables, packed f16 weights ----
    k_convert_f16<<<NB(BL * 256), 256, 0, stream>>>(x, xh, BL * 256);
    k_prep_pph<<<NB(BL * 32), 256, 0, stream>>>(pos_ph, pph, BL * 32);
    k_prep_pos<<<NB((long)Ln * 32), 256, 0, stream>>>(pos_ph, posC, posS, posCt, posSt,
                                                      (long)Ln * 32);

    PW(w_content, WcT, 256, 256, 256, 256, 0);
    PW(w_val,     WvT, 256, 256, 256, 256, 0);
    PW(w_ctx,     WxT, 256, 256, 256, 256, 0);
    PW(w_out,     WoutT, 256, 256, 256, 256, 0);
    PW(w_cpe,     WcpeT, 32, 32, 256, 256, 0);
    PW(sg_w1,     Sg1T, 128, 128, 256, 256, 0);
    PW(kv_w1,     KvW1aT, 256, 256, 256, 512, 0);
    PW(kv_w1,     KvW1bT, 256, 256, 256, 512, 256);
    PW(kv_w2,     KvW2T, 32, 32, 256, 256, 0);
    for (int h = 0; h < Hn; h++) {
        PW(hop_w1 + (size_t)h * 128 * 288, Hop1aT + (size_t)h * 256 * 128, 128, 128, 256, 288, 0);
        PW(hop_w1 + (size_t)h * 128 * 288, Hop1bT + (size_t)h * 32 * 128, 128, 128, 32, 288, 256);
        PW(hop_w2 + (size_t)h * 33 * 128,  Hop2T  + (size_t)h * 128 * 48, 48, 33, 128, 128, 0);
    }

    // ---- dense projections (WMMA) ----
    BI(content, b_content, BL, 256, 256, 256);
    G4(xh, WcT, content, (int)BL, 256, 256, 256, 256);
    k_pack_act<<<NB(BL * 256), 256, 0, stream>>>(content, contenth, 16,
                                                 (long)Ln * 256, BL * 256);

    BI(values, b_val, BL, 256, 256, 256);
    G4(xh, WvT, values, (int)BL, 256, 256, 256, 256);

    BI(ctx, b_ctx, BL, 256, 256, 256);
    G4(xh, WxT, ctx, (int)BL, 256, 256, 256, 256);

    BI(keyph, b_cpe, BL, 32, 32, 32);
    G4(xh, WcpeT, keyph, (int)BL, 32, 256, 256, 32);
    k_tanhpi<<<NB(BL * 32), 256, 0, stream>>>(keyph, BL * 32);

    BI(sg_h, sg_b1, BL, 128, 128, 128);
    G4(xh, Sg1T, sg_h, (int)BL, 128, 256, 256, 128);
    k_gelu<<<NB(BL * 128), 256, 0, stream>>>(sg_h, (half_t*)nullptr, BL * 128);
    k_storegate<<<NB(BL), 256, 0, stream>>>(sg_h, sg_w2, sg_b2, sgate, BL);
    k_gatecum<<<Bn, 32, 0, stream>>>(sgate, gcum);
    k_gatedv<<<NB(BL * 256), 256, 0, stream>>>(values, sgate, gvf, BL * 256);
    k_pack_act<<<NB(BL * 256), 256, 0, stream>>>(gvf, gvh, 16, (long)Ln * 256, BL * 256);

    k_ctxavg<<<NB((long)Bn * 256), 256, 0, stream>>>(ctx, ctxavgh, (long)Bn * 256);

    BI(kvh, kv_b1, BL, 256, 256, 256);
    G4(xh, KvW1aT, kvh, (int)BL, 256, 256, 256, 256);
    G4(ctxavgh, KvW1bT, kvh, (int)BL, 256, 256, 256, 256);
    k_gelu<<<NB(BL * 256), 256, 0, stream>>>(kvh, kvhh, BL * 256);

    BI(sp, kv_b2, BL, 32, 32, 32);
    G4(kvhh, KvW2T, sp, (int)BL, 32, 256, 256, 32);
    k_tanhpi<<<NB(BL * 32), 256, 0, stream>>>(sp, BL * 32);
    k_kva<<<NB(BL * 32), 256, 0, stream>>>(sp, kvCt, kvSt, BL * 32);

    // hop MLP per head (h1 layout (BL, H*128); off layout (BL, H*48))
    for (int h = 0; h < Hn; h++) {
        BI(h1 + h * 128, hop_b1 + (size_t)h * 128, BL, 128, 128, 512);
        G4(xh,  Hop1aT + (size_t)h * 256 * 128, h1 + h * 128, (int)BL, 128, 256, 256, 512);
        G4(pph, Hop1bT + (size_t)h * 32 * 128,  h1 + h * 128, (int)BL, 128, 32,  32,  512);
    }
    k_gelu<<<NB(BL * 512), 256, 0, stream>>>(h1, h1h, BL * 512);
    for (int h = 0; h < Hn; h++) {
        BI(offb + h * 48, hop_b2 + (size_t)h * 33, BL, 48, 33, 192);
        G4(h1h + h * 128, Hop2T + (size_t)h * 128 * 48, offb + h * 48,
           (int)BL, 48, 128, 512, 192);
    }
    k_query<<<NB(BL * 32), 256, 0, stream>>>(offb, posC, posS, QcS, QsS, BL * 32);

    // ---- chunk outer-product sums as batched WMMA GEMMs (P x Lc @ Lc x D) ----
    BI(Sc_ch,  nullptr, (long)Bn * CH * 32, 256, 0, 256);
    BI(Ss_ch,  nullptr, (long)Bn * CH * 32, 256, 0, 256);
    BI(KvC_ch, nullptr, (long)Bn * CH * 32, 256, 0, 256);
    BI(KvS_ch, nullptr, (long)Bn * CH * 32, 256, 0, 256);
    G2(posCt, contenth, Sc_ch, 32, 256, CLEN, Ln, 256, Bn * CH, CH,
       0, CLEN, Ln * 256, CLEN * 256, CH * 8192, 8192);
    G2(posSt, contenth, Ss_ch, 32, 256, CLEN, Ln, 256, Bn * CH, CH,
       0, CLEN, Ln * 256, CLEN * 256, CH * 8192, 8192);
    G2(kvCt, gvh, KvC_ch, 32, 256, CLEN, Ln, 256, Bn * CH, CH,
       32 * Ln, CLEN, Ln * 256, CLEN * 256, CH * 8192, 8192);
    G2(kvSt, gvh, KvS_ch, 32, 256, CLEN, Ln, 256, Bn * CH, CH,
       32 * Ln, CLEN, Ln * 256, CLEN * 256, CH * 8192, 8192);

    long nsc = (long)Bn * 32 * 256;
    k_exclscan<<<NB(nsc), 256, 0, stream>>>(Sc_ch,  CH, nsc);
    k_exclscan<<<NB(nsc), 256, 0, stream>>>(Ss_ch,  CH, nsc);
    k_exclscan<<<NB(nsc), 256, 0, stream>>>(KvC_ch, CH, nsc);
    k_exclscan<<<NB(nsc), 256, 0, stream>>>(KvS_ch, CH, nsc);

    // ---- fused scans: state in registers, queries via LDS ----
    k_mem1scan<<<dim3(CH, Bn), 256, 0, stream>>>(content, posC, posS, QcS, QsS,
                                                 Sc_ch, Ss_ch, combined, Ln, CLEN, CH);
    k_kvscan<<<dim3(CH, Bn), 256, 0, stream>>>(gvf, sp, keyph, gcum,
                                               KvC_ch, KvS_ch, combined, Ln, CLEN, CH);

    // ---- LN + output projection + residual ----
    k_ln<<<(unsigned)BL, 256, 0, stream>>>(combined, ln_g, ln_b, lnh);
    BI(outpre, b_out, BL, 256, 256, 256);
    G4(lnh, WoutT, outpre, (int)BL, 256, 256, 256, 256);
    k_final<<<NB(BL * 256), 256, 0, stream>>>(x, outpre, out, BL * 256);
}
